// TemporalMamba_28449863369507
// MI455X (gfx1250) — compile-verified, matrix operand order checked by reading, not run on hardware
//
#include <hip/hip_runtime.h>
#include <hip/hip_bf16.h>

// ---------------- problem constants ----------------
#define LAYERS 2
#define DM     256
#define DSZ    16
#define DC     4
#define DI     512
#define DR     16
#define BB     4
#define NN     256
#define KK     64
#define MROWS  (BB * NN * KK)        // 65536 rows of (BN,K,*) tensors

typedef _Float16 half8  __attribute__((ext_vector_type(8)));
typedef _Float16 v16h   __attribute__((ext_vector_type(16)));
typedef float    v8f    __attribute__((ext_vector_type(8)));

// ---------------------------------------------------------------------------
// V_WMMA_F32_16X16X32_F16 fragment layout (wave32), per ISA tables:
// lane group g = lane>>4; per lane two contiguous 8-half runs:
//   [k0 + g*8, +8)  and  [k0 + 16 + g*8, +8)
// A uses row m = lane&15; B (32x16) is symmetric with column n = lane&15.
// All runs are 16B aligned for our leading dims (256/512/48/32).
// ---------------------------------------------------------------------------
__device__ __forceinline__ v16h make_frag(half8 lo, half8 hi) {
  return __builtin_shufflevector(lo, hi, 0, 1, 2, 3, 4, 5, 6, 7,
                                 8, 9, 10, 11, 12, 13, 14, 15);
}

// ---------------------------------------------------------------------------
// WMMA GEMM: C[M,N] = A[M,K] * W[N,K]^T   (A,W f16; acc f32)
// Each wave computes a 16 x (NT*16) strip (NT accumulators). KD (the K-dim)
// is a compile-time constant, so the k-loop is FULLY unrolled: straight-line
// code with no loop-carried buffers -> the scheduler sinks loads directly
// into WMMA operand registers (no v_mov rotation) and hoists later steps'
// loads above current WMMAs for latency hiding (partial s_wait_loadcnt).
// N % (NT*16) == 0 and total waves % 8 == 0 by construction -> no bounds
// checks or tail guards: EXEC is all-ones for every WMMA (ISA requirement).
// EPI: 0 = f16 store | 1 = f32 softplus(acc+bias[n]) | 2 = f32 += acc
// ---------------------------------------------------------------------------
template <int EPI, int NT, int KD>
__global__ __launch_bounds__(256) void gemm_wmma_k(
    const _Float16* __restrict__ A, int lda,
    const _Float16* __restrict__ W, int ldw,
    int N, _Float16* __restrict__ outH, float* __restrict__ outF,
    const float* __restrict__ bias, int strips) {
  const int lane = threadIdx.x & 31;
  // wave id is uniform within a wave; readfirstlane pins tile math to SGPRs.
  const int wave = __builtin_amdgcn_readfirstlane(blockIdx.x * 8 + (threadIdx.x >> 5));
  const int mt = wave / strips;
  const int strip = wave - mt * strips;
  const int g = lane >> 4;
  const int l15 = lane & 15;

  // per-lane row base pointers (k-offsets become immediate-offset loads)
  const _Float16* arow = A + (size_t)(mt * 16 + l15) * lda + g * 8;
  const _Float16* wrow[NT];
#pragma unroll
  for (int j = 0; j < NT; ++j)
    wrow[j] = W + (size_t)(strip * (NT * 16) + j * 16 + l15) * ldw + g * 8;

  v8f acc[NT] = {};

#pragma unroll
  for (int k0 = 0; k0 < KD; k0 += 32) {
    // issue the whole step's loads as one clause...
    half8 alo = *reinterpret_cast<const half8*>(arow + k0);
    half8 ahi = *reinterpret_cast<const half8*>(arow + k0 + 16);
    half8 blo[NT], bhi[NT];
#pragma unroll
    for (int j = 0; j < NT; ++j) {
      blo[j] = *reinterpret_cast<const half8*>(wrow[j] + k0);
      bhi[j] = *reinterpret_cast<const half8*>(wrow[j] + k0 + 16);
    }
    // ...then consume with NT back-to-back WMMAs
    const v16h a = make_frag(alo, ahi);
#pragma unroll
    for (int j = 0; j < NT; ++j) {
      const v16h b = make_frag(blo[j], bhi[j]);
      acc[j] = __builtin_amdgcn_wmma_f32_16x16x32_f16(
          false, a, false, b, (short)0, acc[j], false, false);
    }
  }

  // C/D layout: VGPR r -> row m = r + 8*g (within tile), col n = lane&15
#pragma unroll
  for (int j = 0; j < NT; ++j) {
    const int ncol = strip * (NT * 16) + j * 16 + l15;
#pragma unroll
    for (int r = 0; r < 8; ++r) {
      const int row = mt * 16 + r + 8 * g;
      const size_t idx = (size_t)row * N + ncol;
      const float v = acc[j][r];
      if (EPI == 0) {
        outH[idx] = (_Float16)v;
      } else if (EPI == 1) {
        const float t = v + bias[ncol];
        outF[idx] = (t > 20.0f) ? t : log1pf(__expf(t));
      } else {
        outF[idx] += v;
      }
    }
  }
}

// ---------------- layout transforms ----------------
// h[(b*NN+n)*KK + k, :] = x[b, k*NN + n, :]
__global__ __launch_bounds__(256) void reshape_in_k(const float4* __restrict__ x,
                                                    float4* __restrict__ h) {
  const int idx = blockIdx.x * 256 + threadIdx.x;   // over MROWS*64 float4
  const int r = idx >> 6, c = idx & 63;
  const int bn = r >> 6, k = r & 63;
  const int b = bn >> 8, n = bn & 255;
  const int src = (b << 14) + (k << 8) + n;
  h[(size_t)r * 64 + c] = x[(size_t)src * 64 + c];
}

__global__ __launch_bounds__(256) void reshape_out_k(const float4* __restrict__ h,
                                                     float4* __restrict__ out) {
  const int idx = blockIdx.x * 256 + threadIdx.x;
  const int r = idx >> 6, c = idx & 63;
  const int bn = r >> 6, k = r & 63;
  const int b = bn >> 8, n = bn & 255;
  const int dst = (b << 14) + (k << 8) + n;
  out[(size_t)dst * 64 + c] = h[(size_t)r * 64 + c];
}

// ---------------- weight conversion ----------------
__global__ __launch_bounds__(256) void cvt_f16_k(const float* __restrict__ src,
                                                 _Float16* __restrict__ dst, int n) {
  const int i = blockIdx.x * 256 + threadIdx.x;
  if (i < n) dst[i] = (_Float16)src[i];
}

// dt_proj_w (L,DI,DR=16) -> (L,DI,32) zero-padded in k
__global__ __launch_bounds__(256) void pad_dtw_k(const float* __restrict__ src,
                                                 _Float16* __restrict__ dst) {
  const int i = blockIdx.x * 256 + threadIdx.x;    // over LAYERS*DI*32
  if (i >= LAYERS * DI * 32) return;
  const int r = i & 31;
  const int d = (i >> 5) % DI;
  const int l = i / (DI * 32);
  dst[i] = (r < DR) ? (_Float16)src[(l * DI + d) * DR + r] : (_Float16)0.0f;
}

// ---------------- LayerNorm: one wave per row of 256 ----------------
__global__ __launch_bounds__(256) void ln_k(const float* __restrict__ h,
                                            const float* __restrict__ w,
                                            const float* __restrict__ b,
                                            _Float16* __restrict__ out) {
  const int row = blockIdx.x * 8 + (threadIdx.x >> 5);
  const int lane = threadIdx.x & 31;
  const float* p = h + (size_t)row * DM + lane * 8;
  float v[8];
#pragma unroll
  for (int i = 0; i < 8; ++i) v[i] = p[i];
  float s = 0.0f, sq = 0.0f;
#pragma unroll
  for (int i = 0; i < 8; ++i) { s += v[i]; sq += v[i] * v[i]; }
  for (int off = 16; off > 0; off >>= 1) {
    s += __shfl_xor(s, off, 32);
    sq += __shfl_xor(sq, off, 32);
  }
  const float mu = s * (1.0f / DM);
  const float var = sq * (1.0f / DM) - mu * mu;
  const float rs = rsqrtf(var + 1e-5f);
  half8 o;
#pragma unroll
  for (int i = 0; i < 8; ++i) {
    const int c = lane * 8 + i;
    o[i] = (_Float16)((v[i] - mu) * rs * w[c] + b[c]);
  }
  *reinterpret_cast<half8*>(out + (size_t)row * DM + lane * 8) = o;
}

// ---------------- depthwise causal conv (DC=4) + SiLU ----------------
// xin = xz[:, 0:DI]; one block per sequence, thread d walks t.
__global__ __launch_bounds__(512) void conv_silu_k(const _Float16* __restrict__ xz,
                                                   const float* __restrict__ cw,
                                                   const float* __restrict__ cb,
                                                   _Float16* __restrict__ xc) {
  const int bn = blockIdx.x;
  const int d = threadIdx.x;
  const float w0 = cw[d * DC + 0], w1 = cw[d * DC + 1];
  const float w2 = cw[d * DC + 2], w3 = cw[d * DC + 3];
  const float bias = cb[d];
  float p0 = 0.0f, p1 = 0.0f, p2 = 0.0f;
  for (int t = 0; t < KK; ++t) {
    const size_t r = (size_t)bn * KK + t;
    const float xv = (float)xz[r * (2 * DI) + d];
    float a = bias + p0 * w0 + p1 * w1 + p2 * w2 + xv * w3;
    p0 = p1; p1 = p2; p2 = xv;
    a = a / (1.0f + __expf(-a));                   // SiLU
    xc[r * DI + d] = (_Float16)a;
  }
}

// ---------------- selective scan + D-skip + SiLU(z) gate ----------------
// One block per sequence; thread d keeps h[d, 0:16] in registers.
__global__ __launch_bounds__(512) void scan_k(const _Float16* __restrict__ xc,
                                              const float* __restrict__ dt,
                                              const _Float16* __restrict__ dbl,
                                              const _Float16* __restrict__ xz,
                                              const float* __restrict__ A_log,
                                              const float* __restrict__ Dp,
                                              _Float16* __restrict__ y16) {
  const int bn = blockIdx.x;
  const int d = threadIdx.x;
  __shared__ float sB[DSZ];
  __shared__ float sC[DSZ];
  float As[DSZ];
#pragma unroll
  for (int s = 0; s < DSZ; ++s) As[s] = -__expf(A_log[d * DSZ + s]);
  float hst[DSZ];
#pragma unroll
  for (int s = 0; s < DSZ; ++s) hst[s] = 0.0f;
  const float dval = Dp[d];

  for (int t = 0; t < KK; ++t) {
    const size_t r = (size_t)bn * KK + t;
    if (threadIdx.x < 2 * DSZ) {                   // stage B_t, C_t via LDS
      const float v = (float)dbl[r * (DR + 2 * DSZ) + DR + threadIdx.x];
      if (threadIdx.x < DSZ) sB[threadIdx.x] = v;
      else                   sC[threadIdx.x - DSZ] = v;
    }
    __syncthreads();
    const float xv = (float)xc[r * DI + d];
    const float dtv = dt[r * DI + d];
    const float dx = dtv * xv;
    float y = 0.0f;
#pragma unroll
    for (int s = 0; s < DSZ; ++s) {
      const float dA = __expf(dtv * As[s]);
      hst[s] = hst[s] * dA + dx * sB[s];
      y += hst[s] * sC[s];
    }
    const float z = (float)xz[r * (2 * DI) + DI + d];
    float yv = (y + dval * xv) * (z / (1.0f + __expf(-z)));
    y16[r * DI + d] = (_Float16)yv;
    __syncthreads();
  }
}

// ---------------------------------------------------------------------------
extern "C" void kernel_launch(void* const* d_in, const int* in_sizes, int n_in,
                              void* d_out, int out_size, void* d_ws, size_t ws_size,
                              hipStream_t stream) {
  const float* x        = (const float*)d_in[0];
  const float* in_w     = (const float*)d_in[1];   // (L, 2*DI, DM)
  const float* conv_w   = (const float*)d_in[2];   // (L, DI, DC)
  const float* conv_b   = (const float*)d_in[3];   // (L, DI)
  const float* xp_w     = (const float*)d_in[4];   // (L, DR+2*DS, DI)
  const float* dt_w     = (const float*)d_in[5];   // (L, DI, DR)
  const float* dt_b     = (const float*)d_in[6];   // (L, DI)
  const float* A_log    = (const float*)d_in[7];   // (L, DI, DS)
  const float* D_param  = (const float*)d_in[8];   // (L, DI)
  const float* out_w    = (const float*)d_in[9];   // (L, DM, DI)
  const float* ln_w     = (const float*)d_in[10];  // (L, DM)
  const float* ln_b     = (const float*)d_in[11];  // (L, DM)
  float* out = (float*)d_out;

  // --- workspace layout (cursor) ---
  char* ws = (char*)d_ws;
  size_t cur = 0;
  auto alloc = [&](size_t bytes) { void* p = ws + cur; cur += (bytes + 255) & ~(size_t)255; return p; };
  float*     h_buf  = (float*)    alloc((size_t)MROWS * DM * 4);          // residual stream
  _Float16*  hn16   = (_Float16*) alloc((size_t)MROWS * DM * 2);          // LN out
  _Float16*  xz16   = (_Float16*) alloc((size_t)MROWS * 2 * DI * 2);      // in_proj out
  _Float16*  xc16   = (_Float16*) alloc((size_t)MROWS * DI * 2);          // conv+silu out
  _Float16*  dbl16  = (_Float16*) alloc((size_t)MROWS * (DR + 2 * DSZ) * 2);
  float*     dt32   = (float*)    alloc((size_t)MROWS * DI * 4);          // softplus(dt)
  _Float16*  y16    = (_Float16*) alloc((size_t)MROWS * DI * 2);          // scan out
  _Float16*  inw16  = (_Float16*) alloc((size_t)LAYERS * 2 * DI * DM * 2);
  _Float16*  xpw16  = (_Float16*) alloc((size_t)LAYERS * (DR + 2 * DSZ) * DI * 2);
  _Float16*  dtw16  = (_Float16*) alloc((size_t)LAYERS * DI * 32 * 2);    // k-padded
  _Float16*  outw16 = (_Float16*) alloc((size_t)LAYERS * DM * DI * 2);

  // --- 1. reshape input into (BN, K, DM) ---
  {
    const int tot = MROWS * (DM / 4);
    reshape_in_k<<<tot / 256, 256, 0, stream>>>((const float4*)x, (float4*)h_buf);
  }
  // --- 2. weight conversion (once) ---
  {
    int n = LAYERS * 2 * DI * DM;
    cvt_f16_k<<<(n + 255) / 256, 256, 0, stream>>>(in_w, inw16, n);
    n = LAYERS * (DR + 2 * DSZ) * DI;
    cvt_f16_k<<<(n + 255) / 256, 256, 0, stream>>>(xp_w, xpw16, n);
    n = LAYERS * DM * DI;
    cvt_f16_k<<<(n + 255) / 256, 256, 0, stream>>>(out_w, outw16, n);
    n = LAYERS * DI * 32;
    pad_dtw_k<<<(n + 255) / 256, 256, 0, stream>>>(dt_w, dtw16);
  }

  const int TM = MROWS / 16;  // 4096 m-tiles
  for (int l = 0; l < LAYERS; ++l) {
    // --- 3. LayerNorm ---
    ln_k<<<MROWS / 8, 256, 0, stream>>>(h_buf, ln_w + l * DM, ln_b + l * DM, hn16);

    // --- 4. in_proj: xz16 = hn16 (M,256) * in_w^T  (N=1024, NT=4, K=256) ---
    {
      const int N = 2 * DI, strips = N / 64, waves = TM * strips;
      gemm_wmma_k<0, 4, DM><<<waves / 8, 256, 0, stream>>>(
          hn16, DM, inw16 + (size_t)l * 2 * DI * DM, DM, N, xz16, nullptr, nullptr, strips);
    }
    // --- 5. conv + SiLU ---
    conv_silu_k<<<BB * NN, DI, 0, stream>>>(xz16, conv_w + l * DI * DC, conv_b + l * DI, xc16);

    // --- 6. x_proj: dbl16 = xc16 (M,512) * xp_w^T  (N=48, NT=3, K=512) ---
    {
      const int N = DR + 2 * DSZ, strips = N / 48, waves = TM * strips;
      gemm_wmma_k<0, 3, DI><<<waves / 8, 256, 0, stream>>>(
          xc16, DI, xpw16 + (size_t)l * N * DI, DI, N, dbl16, nullptr, nullptr, strips);
    }
    // --- 7. dt_proj: dt32 = softplus(dbl16[:, 0:32] * dtw_pad^T + dt_b)  (N=512, K=32) ---
    {
      const int N = DI, strips = N / 64, waves = TM * strips;
      gemm_wmma_k<1, 4, 32><<<waves / 8, 256, 0, stream>>>(
          dbl16, DR + 2 * DSZ, dtw16 + (size_t)l * DI * 32, 32, N,
          nullptr, dt32, dt_b + l * DI, strips);
    }
    // --- 8. selective scan + gate ---
    scan_k<<<BB * NN, DI, 0, stream>>>(xc16, dt32, dbl16, xz16,
                                       A_log + (size_t)l * DI * DSZ, D_param + l * DI, y16);
    // --- 9. out_proj with residual: h += y16 (M,512) * out_w^T  (N=256, NT=4, K=512) ---
    {
      const int N = DM, strips = N / 64, waves = TM * strips;
      gemm_wmma_k<2, 4, DI><<<waves / 8, 256, 0, stream>>>(
          y16, DI, outw16 + (size_t)l * DM * DI, DI, N, nullptr, h_buf, nullptr, strips);
    }
  }

  // --- 10. reshape back to (B, K*N, DM) ---
  {
    const int tot = MROWS * (DM / 4);
    reshape_out_k<<<tot / 256, 256, 0, stream>>>((const float4*)h_buf, (float4*)out);
  }
}
